// CharDecoder_60026462929171
// MI455X (gfx1250) — compile-verified, matrix operand order checked by reading, not run on hardware
//
#include <hip/hip_runtime.h>
#include <math.h>

// CDNA5 / gfx1250 teacher-forced LSTM decoder.
#define B_  4096
#define T_  16
#define V_  512
#define E_  512
#define H_  1024
#define G_  (4 * H_)        // 4096 gate columns
#define KC_ (E_ + H_)       // 1536 fused K dim ([x_t | h])

typedef __attribute__((ext_vector_type(16))) _Float16 v16h;
typedef __attribute__((ext_vector_type(8)))  _Float16 v8h;
typedef __attribute__((ext_vector_type(8)))  float    v8f;

union F16Frag { v16h v; v8h h[2]; };

// Load one 16x32 (f16) WMMA A/B fragment row for this lane.
// Layout (ISA 7.12.2, 16-bit A 16x32): lanes 0-15 hold K={0..7,16..23},
// lanes 16-31 hold K={8..15,24..31}  ->  kgrp = lane>>4.
__device__ __forceinline__ v16h load_frag16(const _Float16* rowptr, int kgrp) {
    F16Frag u;
    u.h[0] = *(const v8h*)(rowptr + kgrp * 8);
    u.h[1] = *(const v8h*)(rowptr + 16 + kgrp * 8);
    return u.v;
}

__device__ __forceinline__ v8f wmma_f16(v16h a, v16h b, v8f c) {
    return __builtin_amdgcn_wmma_f32_16x16x32_f16(false, a, false, b,
                                                  (short)0, c, false, false);
}

__device__ __forceinline__ float sigmoidf_(float x) {
    return 1.0f / (1.0f + expf(-x));
}

// Software-pipelined 2xM x 4xN GEMM segment over a single contiguous K range.
// Two register fragment sets; loads for step n+1 issue before WMMAs of step n.
template<int KLEN>
__device__ __forceinline__ void gemm_seg_2x4(
    const _Float16* __restrict__ a0p, const _Float16* __restrict__ a1p,
    const _Float16* __restrict__ w0, const _Float16* __restrict__ w1,
    const _Float16* __restrict__ w2, const _Float16* __restrict__ w3,
    int kgrp, v8f acc[2][4])
{
    v16h a0A = load_frag16(a0p, kgrp);
    v16h a1A = load_frag16(a1p, kgrp);
    v16h b0A = load_frag16(w0, kgrp);
    v16h b1A = load_frag16(w1, kgrp);
    v16h b2A = load_frag16(w2, kgrp);
    v16h b3A = load_frag16(w3, kgrp);

    for (int k0 = 0; k0 < KLEN; k0 += 64) {
        const int k1 = k0 + 32;
        v16h a0B = load_frag16(a0p + k1, kgrp);
        v16h a1B = load_frag16(a1p + k1, kgrp);
        v16h b0B = load_frag16(w0 + k1, kgrp);
        v16h b1B = load_frag16(w1 + k1, kgrp);
        v16h b2B = load_frag16(w2 + k1, kgrp);
        v16h b3B = load_frag16(w3 + k1, kgrp);

        acc[0][0] = wmma_f16(a0A, b0A, acc[0][0]);
        acc[0][1] = wmma_f16(a0A, b1A, acc[0][1]);
        acc[0][2] = wmma_f16(a0A, b2A, acc[0][2]);
        acc[0][3] = wmma_f16(a0A, b3A, acc[0][3]);
        acc[1][0] = wmma_f16(a1A, b0A, acc[1][0]);
        acc[1][1] = wmma_f16(a1A, b1A, acc[1][1]);
        acc[1][2] = wmma_f16(a1A, b2A, acc[1][2]);
        acc[1][3] = wmma_f16(a1A, b3A, acc[1][3]);

        if (k0 + 64 < KLEN) {
            const int k2 = k0 + 64;
            a0A = load_frag16(a0p + k2, kgrp);
            a1A = load_frag16(a1p + k2, kgrp);
            b0A = load_frag16(w0 + k2, kgrp);
            b1A = load_frag16(w1 + k2, kgrp);
            b2A = load_frag16(w2 + k2, kgrp);
            b3A = load_frag16(w3 + k2, kgrp);
        }

        acc[0][0] = wmma_f16(a0B, b0B, acc[0][0]);
        acc[0][1] = wmma_f16(a0B, b1B, acc[0][1]);
        acc[0][2] = wmma_f16(a0B, b2B, acc[0][2]);
        acc[0][3] = wmma_f16(a0B, b3B, acc[0][3]);
        acc[1][0] = wmma_f16(a1B, b0B, acc[1][0]);
        acc[1][1] = wmma_f16(a1B, b1B, acc[1][1]);
        acc[1][2] = wmma_f16(a1B, b2B, acc[1][2]);
        acc[1][3] = wmma_f16(a1B, b3B, acc[1][3]);
    }
}

// ---------------------------------------------------------------------------
// Prep: weights -> f16, fuse [W_ih | W_hh] -> Wg, bias = b_ih + b_hh.
// ---------------------------------------------------------------------------
__global__ void k_prep(const float* __restrict__ W_ih, const float* __restrict__ W_hh,
                       const float* __restrict__ W_out, const float* __restrict__ emb,
                       const float* __restrict__ b_ih, const float* __restrict__ b_hh,
                       _Float16* __restrict__ Wg16, _Float16* __restrict__ Wout16,
                       _Float16* __restrict__ emb16, float* __restrict__ bias)
{
    const long stride = (long)gridDim.x * blockDim.x;
    const long tid = (long)blockIdx.x * blockDim.x + threadIdx.x;
    for (long i = tid; i < (long)G_ * KC_; i += stride) {
        long g = i / KC_, k = i % KC_;
        float v = (k < E_) ? W_ih[g * E_ + k] : W_hh[g * H_ + (k - E_)];
        Wg16[i] = (_Float16)v;
    }
    for (long i = tid; i < (long)V_ * H_; i += stride)
        Wout16[i] = (_Float16)W_out[i];
    for (long i = tid; i < (long)V_ * E_; i += stride)
        emb16[i] = (_Float16)emb[i];
    for (long i = tid; i < G_; i += stride)
        bias[i] = b_ih[i] + b_hh[i];
}

// ---------------------------------------------------------------------------
// Init: h0 = c0 = encoded_input; zero per-example loss output.
// ---------------------------------------------------------------------------
__global__ void k_init(const float* __restrict__ enc, _Float16* __restrict__ h16,
                       float* __restrict__ c, float* __restrict__ out)
{
    long i = (long)blockIdx.x * blockDim.x + threadIdx.x;
    if (i < (long)B_ * H_) {
        float v = enc[i];
        h16[i] = (_Float16)v;
        c[i]   = v;
    }
    if (i < B_) out[i] = 0.0f;
}

// ---------------------------------------------------------------------------
// Fused gates GEMM + LSTM cell.
// Block tile 64 rows x 64 j-cols; 8 waves = 2 (M, 32 rows each) x 4 (j).
// Each wave: 2 M-tiles x 4 gate-planes {j,H+j,2H+j,3H+j} = 8 accumulators,
// so i/f/g/o for each (row,col) align elementwise -> in-register cell update.
// K is split statically into the emb segment (512) and the h segment (1024)
// to keep region selects and address recompute out of the hot loop.
// Grid: (H/64, B/64) = (16, 64).
// ---------------------------------------------------------------------------
__global__ __launch_bounds__(256) void k_gates_cell(
    const _Float16* __restrict__ hprev,   // [B,H] f16  (h_t)
    _Float16* __restrict__ hnext,         // [B,H] f16  (h_{t+1})
    float* __restrict__ c,                // [B,H] f32  in-place
    const _Float16* __restrict__ emb16,   // [V,E] f16
    const _Float16* __restrict__ Wg16,    // [4H, KC] f16
    const float* __restrict__ bias,       // [4H]
    const int* __restrict__ target,       // [B,T]
    int t)
{
    const int lane = threadIdx.x & 31;
    const int wave = threadIdx.x >> 5;
    const int kgrp = lane >> 4;
    const int ln   = lane & 15;
    const int mw   = wave >> 2;          // 0..1
    const int jw   = wave & 3;           // 0..3

    const int rowbase = blockIdx.y * 64 + mw * 32;   // 32 batch rows
    const int jbase   = blockIdx.x * 64 + jw * 16;   // 16 hidden cols

    // Two A rows per lane (M-tiles 0 and 1), gathered from emb / hprev.
    const int b0 = rowbase + ln;
    const int b1 = rowbase + 16 + ln;
    const int id0 = (t == 0) ? 0 : target[b0 * T_ + (t - 1)];
    const int id1 = (t == 0) ? 0 : target[b1 * T_ + (t - 1)];
    const _Float16* embrow0 = emb16 + (size_t)id0 * E_;
    const _Float16* embrow1 = emb16 + (size_t)id1 * E_;
    const _Float16* hrow0   = hprev + (size_t)b0  * H_;
    const _Float16* hrow1   = hprev + (size_t)b1  * H_;

    // Four B rows per lane: gate plane g at column jbase+ln.
    const _Float16* w0 = Wg16 + (size_t)(0 * H_ + jbase + ln) * KC_;
    const _Float16* w1 = Wg16 + (size_t)(1 * H_ + jbase + ln) * KC_;
    const _Float16* w2 = Wg16 + (size_t)(2 * H_ + jbase + ln) * KC_;
    const _Float16* w3 = Wg16 + (size_t)(3 * H_ + jbase + ln) * KC_;

    v8f acc[2][4] = {};
    // K segment 1: x_t contribution (emb rows, Wg cols [0,512))
    gemm_seg_2x4<E_>(embrow0, embrow1, w0, w1, w2, w3, kgrp, acc);
    // K segment 2: h contribution (hprev rows, Wg cols [512,1536))
    gemm_seg_2x4<H_>(hrow0, hrow1, w0 + E_, w1 + E_, w2 + E_, w3 + E_, kgrp, acc);

    // Epilogue: bias + LSTM cell in registers.
    // C layout: elem r of lane L -> M = r + 8*(L>>4), N = L&15.
    const int col = jbase + ln;
    const float bia = bias[0 * H_ + col];
    const float bif = bias[1 * H_ + col];
    const float big = bias[2 * H_ + col];
    const float bio = bias[3 * H_ + col];
#pragma unroll
    for (int mi = 0; mi < 2; ++mi) {
        const int rbase = rowbase + mi * 16 + kgrp * 8;
#pragma unroll
        for (int r = 0; r < 8; ++r) {
            const size_t idx = (size_t)(rbase + r) * H_ + col;
            float iv = sigmoidf_(acc[mi][0][r] + bia);
            float fv = sigmoidf_(acc[mi][1][r] + bif);
            float gv = tanhf   (acc[mi][2][r] + big);
            float ov = sigmoidf_(acc[mi][3][r] + bio);
            float cn = fv * c[idx] + iv * gv;
            c[idx] = cn;
            hnext[idx] = (_Float16)(ov * tanhf(cn));
        }
    }
}

// ---------------------------------------------------------------------------
// Fused logits GEMM + logsumexp NLL, software-pipelined.
// Block owns 16 batch rows; 8 waves x 4 N-tiles cover V=512 vocab columns.
// ---------------------------------------------------------------------------
__global__ __launch_bounds__(256) void k_logits_loss(
    const _Float16* __restrict__ h16,     // [B,H] f16 (h_{t+1})
    const _Float16* __restrict__ Wout16,  // [V,H] f16
    const int* __restrict__ target,       // [B,T]
    float* __restrict__ out,              // [B] accumulated loss
    int t)
{
    __shared__ float logits[16][V_];      // 32 KB

    const int lane = threadIdx.x & 31;
    const int wave = threadIdx.x >> 5;
    const int kgrp = lane >> 4;
    const int ln   = lane & 15;
    const int mbase = blockIdx.x * 16;

    const _Float16* arow = h16 + (size_t)(mbase + ln) * H_;
    const int ncol0 = wave * 64;
    const _Float16* w0 = Wout16 + (size_t)(ncol0 +  0 + ln) * H_;
    const _Float16* w1 = Wout16 + (size_t)(ncol0 + 16 + ln) * H_;
    const _Float16* w2 = Wout16 + (size_t)(ncol0 + 32 + ln) * H_;
    const _Float16* w3 = Wout16 + (size_t)(ncol0 + 48 + ln) * H_;

    v8f acc[4] = {};
    v16h aA, b0A, b1A, b2A, b3A;

    aA  = load_frag16(arow, kgrp);
    b0A = load_frag16(w0, kgrp);
    b1A = load_frag16(w1, kgrp);
    b2A = load_frag16(w2, kgrp);
    b3A = load_frag16(w3, kgrp);

    for (int k0 = 0; k0 < H_; k0 += 64) {
        const int k1 = k0 + 32;
        v16h aB  = load_frag16(arow + k1, kgrp);
        v16h b0B = load_frag16(w0 + k1, kgrp);
        v16h b1B = load_frag16(w1 + k1, kgrp);
        v16h b2B = load_frag16(w2 + k1, kgrp);
        v16h b3B = load_frag16(w3 + k1, kgrp);

        acc[0] = wmma_f16(aA, b0A, acc[0]);
        acc[1] = wmma_f16(aA, b1A, acc[1]);
        acc[2] = wmma_f16(aA, b2A, acc[2]);
        acc[3] = wmma_f16(aA, b3A, acc[3]);

        if (k0 + 64 < H_) {
            const int k2 = k0 + 64;
            aA  = load_frag16(arow + k2, kgrp);
            b0A = load_frag16(w0 + k2, kgrp);
            b1A = load_frag16(w1 + k2, kgrp);
            b2A = load_frag16(w2 + k2, kgrp);
            b3A = load_frag16(w3 + k2, kgrp);
        }

        acc[0] = wmma_f16(aB, b0B, acc[0]);
        acc[1] = wmma_f16(aB, b1B, acc[1]);
        acc[2] = wmma_f16(aB, b2B, acc[2]);
        acc[3] = wmma_f16(aB, b3B, acc[3]);
    }

#pragma unroll
    for (int nt = 0; nt < 4; ++nt) {
        const int col = ncol0 + nt * 16 + ln;
#pragma unroll
        for (int r = 0; r < 8; ++r)
            logits[r + kgrp * 8][col] = acc[nt][r];
    }
    __syncthreads();

    // Each wave reduces 2 rows (8 waves x 2 = 16 rows).
    for (int rr = wave * 2; rr < wave * 2 + 2; ++rr) {
        float mx = -3.402823466e38f;
        for (int cidx = lane; cidx < V_; cidx += 32)
            mx = fmaxf(mx, logits[rr][cidx]);
        for (int off = 16; off > 0; off >>= 1)
            mx = fmaxf(mx, __shfl_xor(mx, off, 32));
        float s = 0.0f;
        for (int cidx = lane; cidx < V_; cidx += 32)
            s += expf(logits[rr][cidx] - mx);
        for (int off = 16; off > 0; off >>= 1)
            s += __shfl_xor(s, off, 32);
        if (lane == 0) {
            const int bb = mbase + rr;
            const int tg = target[bb * T_ + t];
            const float loss = (mx + logf(s)) - logits[rr][tg];
            out[bb] += loss * (1.0f / (float)T_);
        }
    }
}

// ---------------------------------------------------------------------------
// Host launcher. Workspace: Wg16 | Wout16 | emb16 | bias | h16 x2 | c  (~48 MB)
// ---------------------------------------------------------------------------
extern "C" void kernel_launch(void* const* d_in, const int* in_sizes, int n_in,
                              void* d_out, int out_size, void* d_ws, size_t ws_size,
                              hipStream_t stream)
{
    const float* enc    = (const float*)d_in[0];
    const int*   target = (const int*)  d_in[1];
    const float* emb    = (const float*)d_in[2];
    const float* W_ih   = (const float*)d_in[3];
    const float* W_hh   = (const float*)d_in[4];
    const float* b_ih   = (const float*)d_in[5];
    const float* b_hh   = (const float*)d_in[6];
    const float* W_out  = (const float*)d_in[7];
    float* out = (float*)d_out;

    char* ws = (char*)d_ws;
    size_t o = 0;
    _Float16* Wg16   = (_Float16*)(ws + o); o += (size_t)G_ * KC_ * 2;
    _Float16* Wout16 = (_Float16*)(ws + o); o += (size_t)V_ * H_ * 2;
    _Float16* emb16  = (_Float16*)(ws + o); o += (size_t)V_ * E_ * 2;
    float*    bias   = (float*)   (ws + o); o += (size_t)G_ * 4;
    _Float16* hbuf0  = (_Float16*)(ws + o); o += (size_t)B_ * H_ * 2;
    _Float16* hbuf1  = (_Float16*)(ws + o); o += (size_t)B_ * H_ * 2;
    float*    c      = (float*)   (ws + o); o += (size_t)B_ * H_ * 4;

    k_prep<<<4096, 256, 0, stream>>>(W_ih, W_hh, W_out, emb, b_ih, b_hh,
                                     Wg16, Wout16, emb16, bias);
    k_init<<<(B_ * H_ + 255) / 256, 256, 0, stream>>>(enc, hbuf0, c, out);

    _Float16* hb[2] = {hbuf0, hbuf1};
    for (int t = 0; t < T_; ++t) {
        _Float16* hprev = hb[t & 1];
        _Float16* hnext = hb[(t + 1) & 1];
        dim3 gg(H_ / 64, B_ / 64);                 // (16, 64) blocks
        k_gates_cell<<<gg, 256, 0, stream>>>(hprev, hnext, c, emb16, Wg16,
                                             bias, target, t);
        k_logits_loss<<<B_ / 16, 256, 0, stream>>>(hnext, Wout16, target, out, t);
    }
}
